// TemporalSetsGNN_89421219103482
// MI455X (gfx1250) — compile-verified, matrix operand order checked by reading, not run on hardware
//
#include <hip/hip_runtime.h>
#include <hip/hip_bf16.h>
#include <math.h>

#define EMB   64
#define BATCH 32
#define NS    5
#define NM    25
#define NP    125     // NS*NM
#define PPAD  128
#define HDIM  192
#define H1    128
#define H2DIM 64

#ifndef __has_builtin
#define __has_builtin(x) 0
#endif
#if __has_builtin(__builtin_amdgcn_global_load_async_to_lds_b128)
#define HAVE_ASYNC_LDS 1
#else
#define HAVE_ASYNC_LDS 0
#endif

typedef __attribute__((ext_vector_type(16))) _Float16 v16h;
typedef __attribute__((ext_vector_type(8)))  _Float16 v8h;
typedef __attribute__((ext_vector_type(8)))  float    v8f;
typedef int v4i_gcc __attribute__((vector_size(16)));   // matches builtin param type

__device__ __forceinline__ void wait_async0() {
#if HAVE_ASYNC_LDS
#if __has_builtin(__builtin_amdgcn_s_wait_asynccnt)
  __builtin_amdgcn_s_wait_asynccnt(0);
#else
  asm volatile("s_wait_asynccnt 0" ::: "memory");
#endif
#endif
}

// Async-DMA one 16-byte beat of a contiguous global slab into LDS
// (GLOBAL_LOAD_ASYNC_TO_LDS_B128, tracked by ASYNCcnt).  The builtin takes
// generic int4* pointers (per the clang diagnostic); address-space inference
// recovers the global/LDS operands.
__device__ __forceinline__ void async_copy_b128(const float* gsrc, float* lds) {
#if HAVE_ASYNC_LDS
  __builtin_amdgcn_global_load_async_to_lds_b128((v4i_gcc*)gsrc, (v4i_gcc*)lds,
                                                 0, 0);
#else
  (void)gsrc; (void)lds;
#endif
}

// Load one 16x32 f16 WMMA operand fragment from LDS (row-major, `stride`
// halfwords per row).  Per the CDNA5 ISA A-matrix layout, each lane needs two
// contiguous runs of 8 halves: k = hi*8 + [0,8) and k = 16 + hi*8 + [0,8)
// (hi = lane>>4), i.e. two 16-byte ds loads.  The B operand uses the
// symmetric layout, so the same loader works on an n-major (transposed)
// weight tile.
__device__ __forceinline__ v16h lds_frag(const _Float16* base, int tileRow,
                                         int stride, int kbase, int lane) {
  const int r  = tileRow + (lane & 15);
  const int hi = lane >> 4;
  const _Float16* p = base + r * stride + kbase + hi * 8;
  v8h lo = *(const v8h*)p;
  v8h hh = *(const v8h*)(p + 16);
  v16h out;
#pragma unroll
  for (int i = 0; i < 8; ++i) { out[i] = lo[i]; out[i + 8] = hh[i]; }
  return out;
}

// Sparse GCN aggregation: agg[p] = (sum_s mult[p][s]*showSum[s] - dup*x[p]) /
// (den[p] + 1e-10).  Exactly reproduces row-normalized co-occurrence @ x,
// including duplicate song ids within/across shows.
__device__ void gcn_aggregate(const _Float16* xin, const int* ids,
                              const unsigned char* mult, const float* den,
                              float* ssum, _Float16* aggh, int tid) {
  for (int idx = tid; idx < NS * EMB; idx += 256) {
    int s = idx >> 6, d = idx & 63;
    float acc = 0.f;
    for (int j = 0; j < NM; ++j) {
      int p = s * NM + j;
      if (ids[p] > 0) acc += (float)xin[p * EMB + d];
    }
    ssum[idx] = acc;
  }
  __syncthreads();
  for (int idx = tid; idx < PPAD * EMB; idx += 256) {
    int p = idx >> 6, d = idx & 63;
    float nm = 0.f, dup = 0.f;
#pragma unroll
    for (int s = 0; s < NS; ++s) {
      float mm = (float)mult[p * 8 + s];
      nm  += mm * ssum[s * EMB + d];
      dup += mm;
    }
    nm -= dup * (float)xin[idx];
    aggh[idx] = (_Float16)(nm / (den[p] + 1e-10f));
  }
}

// [128x64] x [64x64] -> relu(acc + bias) -> f16 out.  8 waves x 4 tiles
// cover the 8x4 tile grid; 2 K-steps of V_WMMA_F32_16X16X32_F16 each.
__device__ void gemm_relu_f16out(const _Float16* A, const _Float16* Wt,
                                 const float* bias, _Float16* out,
                                 int wv, int lane) {
#pragma unroll
  for (int j = 0; j < 4; ++j) {
    int tile = wv + 8 * j;           // 0..31
    int mt = tile & 7, nt = tile >> 3;
    v8f acc = {};
#pragma unroll
    for (int kt = 0; kt < 2; ++kt) {
      v16h a = lds_frag(A,  mt * 16, EMB, kt * 32, lane);
      v16h b = lds_frag(Wt, nt * 16, EMB, kt * 32, lane);
      acc = __builtin_amdgcn_wmma_f32_16x16x32_f16(false, a, false, b,
                                                   (short)0, acc, false, false);
    }
    int n = nt * 16 + (lane & 15);
    float bv = bias[n];
#pragma unroll
    for (int r = 0; r < 8; ++r) {
      int m = mt * 16 + r + (lane >> 4) * 8;   // D layout: VGPR r -> M=r / r+8
      float v = acc[r] + bv;
      out[m * EMB + n] = (_Float16)(v > 0.f ? v : 0.f);
    }
  }
}

__device__ void load_wt_f16(const float* w, _Float16* wt, int tid) {
  // wt[n][k] = w[k][n]  (n-major so B fragments are contiguous in k)
  for (int idx = tid; idx < EMB * EMB; idx += 256) {
    int n = idx >> 6, k = idx & 63;
    wt[idx] = (_Float16)w[k * EMB + n];
  }
}

__global__ void __launch_bounds__(256) gnn_attn_kernel(
    const int* song_ids, const int* prev,
    const int* venue_ids, const int* tour_ids, const int* country_ids,
    const int* is_fest, const int* is_mara,
    const float* song_emb,
    const float* gcn_w1, const float* gcn_b1,
    const float* gcn_w2, const float* gcn_b2,
    const float* q_w, const float* q_b,
    const float* k_w, const float* k_b,
    const float* v_w, const float* v_b,
    const float* venue_t, const float* tour_t, const float* country_t,
    const float* fest_t, const float* mara_t,
    const float* ctx_w, const float* ctx_b,
    float* h_out /* [BATCH, 192] */) {
  __shared__ int   s_ids[PPAD];
  __shared__ float s_cnt[NS];
  __shared__ unsigned char s_mult[PPAD * 8];
  __shared__ float s_den[PPAD];
  __shared__ _Float16 __attribute__((aligned(16))) s_xb[PPAD * EMB];   // x0/x1/x2
  __shared__ _Float16 __attribute__((aligned(16))) s_aggh[PPAD * EMB];
  __shared__ _Float16 __attribute__((aligned(16))) s_wt[EMB * EMB];
  __shared__ float s_ssum[NS * EMB];
  __shared__ float s_show[NS * EMB];
  __shared__ float s_ctxin[64];
  __shared__ float s_ctx[EMB];
  __shared__ float s_q[EMB];
  __shared__ float s_kk[NS * EMB];
  __shared__ float s_vv[NS * EMB];
  __shared__ float s_att[NS];
  __shared__ int   s_cand;

  const int b    = blockIdx.x;
  const int tid  = threadIdx.x;
  const int lane = tid & 31;
  const int wv   = tid >> 5;

  // Warm GL2 with the weights needed after the sparse-aggregation phase
  // (global_prefetch_b8; fire-and-forget, no counter traffic).
  if (tid < 128) {
    __builtin_prefetch((const char*)gcn_w2 + tid * 128, 0, 1);
    __builtin_prefetch((const char*)k_w    + tid * 128, 0, 1);
    __builtin_prefetch((const char*)v_w    + tid * 128, 0, 1);
    __builtin_prefetch((const char*)q_w    + tid * 128, 0, 1);
  }

  if (tid < PPAD) s_ids[tid] = (tid < NP) ? prev[b * NP + tid] : 0;
  if (tid >= 128 && tid < 128 + NS) {
    int s = tid - 128;
    float c = 0.f;
    for (int j = 0; j < NM; ++j) c += (prev[b * NP + s * NM + j] > 0) ? 1.f : 0.f;
    s_cnt[s] = c;
  }
  __syncthreads();

  // per-slot multiplicity per show (duplicate handling) + row degree
  if (tid < PPAD) {
    const int id = s_ids[tid];
    float den = 0.f;
#pragma unroll
    for (int s = 0; s < NS; ++s) {
      int mm = 0;
      if (id > 0)
        for (int j = 0; j < NM; ++j) mm += (s_ids[s * NM + j] == id) ? 1 : 0;
      s_mult[tid * 8 + s] = (unsigned char)mm;
      den += (float)mm * (s_cnt[s] - 1.f);
    }
    s_den[tid] = den;
  }
  // gather x0 = song_emb[ids]  (padding id 0 -> zero row by construction)
  for (int idx = tid; idx < PPAD * EMB; idx += 256) {
    int p = idx >> 6, d = idx & 63;
    s_xb[idx] = (_Float16)song_emb[s_ids[p] * EMB + d];
  }
  load_wt_f16(gcn_w1, s_wt, tid);
  __syncthreads();

  // ---- GCN layer 1 ----
  gcn_aggregate(s_xb, s_ids, s_mult, s_den, s_ssum, s_aggh, tid);
  __syncthreads();
  gemm_relu_f16out(s_aggh, s_wt, gcn_b1, s_xb, wv, lane);   // s_xb <- x1
  __syncthreads();

  // ---- GCN layer 2 ----
  load_wt_f16(gcn_w2, s_wt, tid);
  gcn_aggregate(s_xb, s_ids, s_mult, s_den, s_ssum, s_aggh, tid);
  __syncthreads();
  gemm_relu_f16out(s_aggh, s_wt, gcn_b2, s_xb, wv, lane);   // s_xb <- x2
  __syncthreads();

  // masked-mean show embeddings
  for (int idx = tid; idx < NS * EMB; idx += 256) {
    int s = idx >> 6, d = idx & 63;
    float acc = 0.f;
    for (int j = 0; j < NM; ++j) {
      int p = s * NM + j;
      if (s_ids[p] > 0) acc += (float)s_xb[p * EMB + d];
    }
    float c = s_cnt[s];
    s_show[idx] = (c > 0.f) ? (acc / c) : 0.f;
  }
  // context feature gather (56 dims)
  if (tid >= 192 && tid < 192 + 56) {
    int i = tid - 192;
    float v;
    if (i < 16)      v = venue_t[venue_ids[b] * 16 + i];
    else if (i < 32) v = tour_t[tour_ids[b] * 16 + (i - 16)];
    else if (i < 40) v = country_t[country_ids[b] * 8 + (i - 32)];
    else if (i < 48) v = fest_t[is_fest[b] * 8 + (i - 40)];
    else             v = mara_t[is_mara[b] * 8 + (i - 48)];
    s_ctxin[i] = v;
  }
  // candidate slot (not found => zero adjacency row => relu(b2))
  if (tid == 0) {
    const int sid = song_ids[b];
    int ci = -1;
    for (int p = 0; p < PPAD; ++p)
      if (s_ids[p] == sid) { ci = p; break; }
    s_cand = ci;
  }
  __syncthreads();

  if (tid < EMB) {
    float acc = ctx_b[tid];
    for (int i = 0; i < 56; ++i) acc += s_ctxin[i] * ctx_w[i * EMB + tid];
    s_ctx[tid] = acc;
    h_out[b * HDIM + EMB + tid] = acc;                    // ctx slice
    float c = (s_cand >= 0) ? (float)s_xb[s_cand * EMB + tid]
                            : fmaxf(gcn_b2[tid], 0.f);
    h_out[b * HDIM + tid] = c;                            // cand slice
  }
  // K / V projections of show embeddings
  for (int idx = tid; idx < NS * EMB; idx += 256) {
    int s = idx >> 6, d = idx & 63;
    float ak = k_b[d], av = v_b[d];
    for (int k = 0; k < EMB; ++k) {
      float sv = s_show[s * EMB + k];
      ak += sv * k_w[k * EMB + d];
      av += sv * v_w[k * EMB + d];
    }
    s_kk[idx] = ak;
    s_vv[idx] = av;
  }
  __syncthreads();
  if (tid < EMB) {
    float acc = q_b[tid];
    for (int k = 0; k < EMB; ++k) acc += s_ctx[k] * q_w[k * EMB + tid];
    s_q[tid] = acc;
  }
  __syncthreads();
  if (tid < NS) {
    float sc = 0.f;
    for (int d = 0; d < EMB; ++d) sc += s_q[d] * s_kk[tid * EMB + d];
    s_att[tid] = sc * 0.125f;   // / sqrt(64)
  }
  __syncthreads();
  if (tid == 0) {
    float mx = s_att[0];
    for (int s = 1; s < NS; ++s) mx = fmaxf(mx, s_att[s]);
    float sum = 0.f;
    for (int s = 0; s < NS; ++s) { float e = expf(s_att[s] - mx); s_att[s] = e; sum += e; }
    float inv = 1.f / sum;
    for (int s = 0; s < NS; ++s) s_att[s] *= inv;
  }
  __syncthreads();
  if (tid < EMB) {
    float acc = 0.f;
#pragma unroll
    for (int s = 0; s < NS; ++s) acc += s_att[s] * s_vv[s * EMB + tid];
    h_out[b * HDIM + 2 * EMB + tid] = acc;                // attended slice
  }
}

__global__ void __launch_bounds__(256) predictor_kernel(
    const float* h, const float* p1_w, const float* p1_b,
    const float* p2_w, const float* p2_b,
    const float* p3_w, const float* p3_b, float* out) {
  __shared__ _Float16 __attribute__((aligned(16))) s_h[BATCH * HDIM];  // 12KB
  __shared__ float    __attribute__((aligned(16))) s_stage[32 * H1];   // 16KB f32 K-chunk (async staged)
  __shared__ _Float16 __attribute__((aligned(16))) s_wc[H1 * 32];      // 8KB  n-major f16 chunk
  __shared__ _Float16 __attribute__((aligned(16))) s_h1[BATCH * H1];   // 8KB
  __shared__ float s_h2[BATCH * H2DIM];                                // 8KB
  const int tid = threadIdx.x, lane = tid & 31, wv = tid >> 5;

  for (int idx = tid; idx < BATCH * HDIM; idx += 256)
    s_h[idx] = (_Float16)h[idx];
  __syncthreads();

  // h @ p1 : [32x192]x[192x128], 2x8 tile grid, 2 tiles/wave.  Each K-chunk
  // of p1_w is a contiguous 16KB slab -> async DMA global->LDS, then
  // transpose-convert to the n-major f16 tile the WMMA B operand wants.
  v8f acc[2] = {};
  for (int kt = 0; kt < 6; ++kt) {
#if HAVE_ASYNC_LDS
    {
      const float* gsrc = p1_w + kt * 32 * H1;
      for (int i = tid; i < (32 * H1) / 4; i += 256)
        async_copy_b128(gsrc + i * 4, s_stage + i * 4);
      wait_async0();
    }
#else
    for (int i = tid; i < 32 * H1; i += 256)
      s_stage[i] = p1_w[kt * 32 * H1 + i];
#endif
    __syncthreads();
    for (int idx = tid; idx < H1 * 32; idx += 256) {
      int n = idx >> 5, kk = idx & 31;
      s_wc[idx] = (_Float16)s_stage[kk * H1 + n];
    }
    __syncthreads();
#pragma unroll
    for (int j = 0; j < 2; ++j) {
      int tile = wv + 8 * j;            // 0..15
      int mt = tile & 1, nt = tile >> 1;
      v16h a = lds_frag(s_h,  mt * 16, HDIM, kt * 32, lane);
      v16h b = lds_frag(s_wc, nt * 16, 32, 0, lane);
      acc[j] = __builtin_amdgcn_wmma_f32_16x16x32_f16(false, a, false, b,
                                                      (short)0, acc[j], false, false);
    }
    __syncthreads();
  }
#pragma unroll
  for (int j = 0; j < 2; ++j) {
    int tile = wv + 8 * j;
    int mt = tile & 1, nt = tile >> 1;
    int n = nt * 16 + (lane & 15);
    float bv = p1_b[n];
#pragma unroll
    for (int r = 0; r < 8; ++r) {
      int m = mt * 16 + r + (lane >> 4) * 8;
      float v = acc[j][r] + bv;
      s_h1[m * H1 + n] = (_Float16)(v > 0.f ? v : 0.f);
    }
  }

  // h1 @ p2 : [32x128]x[128x64], 2x4 tile grid, 1 tile/wave, K chunked by 32
  // with the same async-staged weight path (8KB contiguous slabs of p2_w).
  {
    int mt = wv & 1, nt = wv >> 1;
    v8f a2 = {};
    for (int kt = 0; kt < 4; ++kt) {
#if HAVE_ASYNC_LDS
      {
        const float* gsrc = p2_w + kt * 32 * H2DIM;
        for (int i = tid; i < (32 * H2DIM) / 4; i += 256)
          async_copy_b128(gsrc + i * 4, s_stage + i * 4);
        wait_async0();
      }
#else
      for (int i = tid; i < 32 * H2DIM; i += 256)
        s_stage[i] = p2_w[kt * 32 * H2DIM + i];
#endif
      __syncthreads();
      for (int idx = tid; idx < H2DIM * 32; idx += 256) {
        int n = idx >> 5, kk = idx & 31;
        s_wc[idx] = (_Float16)s_stage[kk * H2DIM + n];
      }
      __syncthreads();
      v16h a = lds_frag(s_h1, mt * 16, H1, kt * 32, lane);
      v16h b = lds_frag(s_wc, nt * 16, 32, 0, lane);
      a2 = __builtin_amdgcn_wmma_f32_16x16x32_f16(false, a, false, b,
                                                  (short)0, a2, false, false);
      __syncthreads();
    }
    int n = nt * 16 + (lane & 15);
    float bv = p2_b[n];
#pragma unroll
    for (int r = 0; r < 8; ++r) {
      int m = mt * 16 + r + (lane >> 4) * 8;
      float v = a2[r] + bv;
      s_h2[m * H2DIM + n] = v > 0.f ? v : 0.f;
    }
  }
  __syncthreads();

  if (tid < BATCH) {
    float acc3 = p3_b[0];
    for (int d = 0; d < H2DIM; ++d) acc3 += s_h2[tid * H2DIM + d] * p3_w[d];
    out[tid] = 1.f / (1.f + expf(-acc3));
  }
}

extern "C" void kernel_launch(void* const* d_in, const int* in_sizes, int n_in,
                              void* d_out, int out_size, void* d_ws, size_t ws_size,
                              hipStream_t stream) {
  (void)in_sizes; (void)n_in; (void)out_size; (void)ws_size;
  const int* song_ids    = (const int*)d_in[0];
  const int* prev        = (const int*)d_in[1];
  const int* venue_ids   = (const int*)d_in[2];
  const int* tour_ids    = (const int*)d_in[3];
  const int* country_ids = (const int*)d_in[4];
  const int* is_fest     = (const int*)d_in[5];
  const int* is_mara     = (const int*)d_in[6];
  const float* song_emb  = (const float*)d_in[7];
  const float* gcn_w1    = (const float*)d_in[8];
  const float* gcn_b1    = (const float*)d_in[9];
  const float* gcn_w2    = (const float*)d_in[10];
  const float* gcn_b2    = (const float*)d_in[11];
  const float* q_w       = (const float*)d_in[12];
  const float* q_b       = (const float*)d_in[13];
  const float* k_w       = (const float*)d_in[14];
  const float* k_b       = (const float*)d_in[15];
  const float* v_w       = (const float*)d_in[16];
  const float* v_b       = (const float*)d_in[17];
  const float* venue_t   = (const float*)d_in[18];
  const float* tour_t    = (const float*)d_in[19];
  const float* country_t = (const float*)d_in[20];
  const float* fest_t    = (const float*)d_in[21];
  const float* mara_t    = (const float*)d_in[22];
  const float* ctx_w     = (const float*)d_in[23];
  const float* ctx_b     = (const float*)d_in[24];
  const float* p1_w      = (const float*)d_in[25];
  const float* p1_b      = (const float*)d_in[26];
  const float* p2_w      = (const float*)d_in[27];
  const float* p2_b      = (const float*)d_in[28];
  const float* p3_w      = (const float*)d_in[29];
  const float* p3_b      = (const float*)d_in[30];
  float* h = (float*)d_ws;   // [BATCH, 192] f32

  gnn_attn_kernel<<<BATCH, 256, 0, stream>>>(
      song_ids, prev, venue_ids, tour_ids, country_ids, is_fest, is_mara,
      song_emb, gcn_w1, gcn_b1, gcn_w2, gcn_b2, q_w, q_b, k_w, k_b, v_w, v_b,
      venue_t, tour_t, country_t, fest_t, mara_t, ctx_w, ctx_b, h);
  predictor_kernel<<<1, 256, 0, stream>>>(h, p1_w, p1_b, p2_w, p2_b,
                                          p3_w, p3_b, (float*)d_out);
}
